// NonParametricProtoCritic_42399917146659
// MI455X (gfx1250) — matrix operand order repaired and checked by reading, not compiled, not run on hardware
//
#include <hip/hip_runtime.h>
#include <stdint.h>

typedef __attribute__((ext_vector_type(16))) __bf16 v16bf;
typedef __attribute__((ext_vector_type(8)))  __bf16 v8bf;
typedef __attribute__((ext_vector_type(8)))  float  v8f;

#define BATCH   1024
#define OBSD    512
#define ACTD    64
#define KD      128
#define CAP     131072
#define NCHUNK  128          // key chunks (one per blockIdx.x of k_scores)
#define CHUNKK  (CAP / NCHUNK)   // 1024 keys per chunk
#define SUBK    64           // keys staged in LDS per sub-iteration
#define MTILE   128          // batch rows per block (8 waves x 16)
#define LPAD    8            // LDS row padding (bf16 elems) -> 272B stride, conflict-free b128

#define EXP2F(x) __builtin_amdgcn_exp2f(x)   // raw v_exp_f32

__device__ __forceinline__ uint32_t bf16pk(float a, float b) {
    uint32_t ua = __float_as_uint(a), ub = __float_as_uint(b);
    ua = (ua + 0x7FFFu + ((ua >> 16) & 1u)) >> 16;   // RNE fp32 -> bf16
    ub = (ub + 0x7FFFu + ((ub >> 16) & 1u)) >> 16;
    return ua | (ub << 16);
}

__device__ __forceinline__ void lse_merge(float& m, float& d, float& n,
                                          float mo, float dm, float nm) {
    float mn = fmaxf(m, mo);
    float s  = EXP2F(m - mn);
    float so = EXP2F(mo - mn);
    d = d * s + dm * so;
    n = n * s + nm * so;
    m = mn;
}

// ---------------------------------------------------------------------------
// K1: predictor GEMV + L2 normalize + trunk MLP (relu, linear).
// Output h is pre-scaled by (1/T)*log2(e) and stored as bf16 for the WMMA GEMM.
// ---------------------------------------------------------------------------
__global__ __launch_bounds__(128)
void k_trunk(const float* __restrict__ obs, const float* __restrict__ action,
             const float* __restrict__ pW, const float* __restrict__ pb,
             const float* __restrict__ w1, const float* __restrict__ b1,
             const float* __restrict__ w2, const float* __restrict__ b2,
             const float* __restrict__ log_temp, unsigned short* __restrict__ hb) {
    __shared__ float sObs[OBSD];
    __shared__ float sH0[KD + ACTD];
    __shared__ float sH1[KD];
    __shared__ float sX[KD];
    const int row = blockIdx.x, t = threadIdx.x;

    for (int i = t; i < OBSD; i += 128) sObs[i] = obs[(size_t)row * OBSD + i];
    if (t < ACTD) sH0[KD + t] = action[(size_t)row * ACTD + t];
    __syncthreads();

    // predictor: inpt[t] = obs . pW[t,:] + pb[t]
    float acc = pb[t];
    const float* wr = pW + (size_t)t * OBSD;
    #pragma unroll 4
    for (int d = 0; d < OBSD; d += 4) {
        float4 w = *(const float4*)(wr + d);
        acc = fmaf(w.x, sObs[d + 0], acc);
        acc = fmaf(w.y, sObs[d + 1], acc);
        acc = fmaf(w.z, sObs[d + 2], acc);
        acc = fmaf(w.w, sObs[d + 3], acc);
    }
    sX[t] = acc;
    __syncthreads();

    float ss = 0.f;
    for (int i = 0; i < KD; ++i) ss = fmaf(sX[i], sX[i], ss);
    float inv = 1.f / fmaxf(sqrtf(ss), 1e-12f);
    sH0[t] = acc * inv;
    __syncthreads();

    // layer 1 + relu
    float a1 = b1[t];
    const float* w1r = w1 + (size_t)t * (KD + ACTD);
    #pragma unroll 4
    for (int j = 0; j < KD + ACTD; j += 4) {
        float4 w = *(const float4*)(w1r + j);
        a1 = fmaf(w.x, sH0[j + 0], a1);
        a1 = fmaf(w.y, sH0[j + 1], a1);
        a1 = fmaf(w.z, sH0[j + 2], a1);
        a1 = fmaf(w.w, sH0[j + 3], a1);
    }
    sH1[t] = fmaxf(a1, 0.f);
    __syncthreads();

    // layer 2
    float a2 = b2[t];
    const float* w2r = w2 + (size_t)t * KD;
    #pragma unroll 4
    for (int j = 0; j < KD; j += 4) {
        float4 w = *(const float4*)(w2r + j);
        a2 = fmaf(w.x, sH1[j + 0], a2);
        a2 = fmaf(w.y, sH1[j + 1], a2);
        a2 = fmaf(w.z, sH1[j + 2], a2);
        a2 = fmaf(w.w, sH1[j + 3], a2);
    }
    // fold 1/T and log2(e) so the softmax needs only exp2 per score
    float hscale = __expf(-log_temp[0]) * 1.44269504088896340736f;
    float hv = a2 * hscale;
    uint32_t u = __float_as_uint(hv);
    u = (u + 0x7FFFu + ((u >> 16) & 1u)) >> 16;
    hb[(size_t)row * KD + t] = (unsigned short)u;
}

// ---------------------------------------------------------------------------
// K2: streaming scores GEMM (bf16 WMMA) fused with online softmax + value sum.
// Grid: (NCHUNK key-chunks, BATCH/MTILE batch tiles). 256 thr = 8 waves.
// ---------------------------------------------------------------------------
__global__ __launch_bounds__(256)
void k_scores(const float* __restrict__ keys, const float* __restrict__ values,
              const unsigned short* __restrict__ hb, float* __restrict__ partials) {
    __shared__ unsigned short sH[MTILE][KD + LPAD];  // 34.8 KB
    __shared__ unsigned short sK[SUBK][KD + LPAD];   // 17.4 KB
    __shared__ float sV[SUBK];

    const int t = threadIdx.x;
    const int lane = t & 31, wave = t >> 5;
    const int mBase = blockIdx.y * MTILE;
    const int kChunk0 = blockIdx.x * CHUNKK;

    // stage H tile (already bf16): 128x128 -> 2048 uint4, 8 per thread
    #pragma unroll
    for (int i = 0; i < 8; ++i) {
        int idx = t + i * 256;
        int r = idx >> 4, c = (idx & 15) << 3;
        *(uint4*)&sH[r][c] = *(const uint4*)&hb[(size_t)(mBase + r) * KD + c];
    }
    __syncthreads();

    // A fragments (16x32 bf16, ISA 7.12.2 layout), reused for entire chunk
    const int arow = (wave << 4) + (lane & 15);
    const int koff = (lane >> 4) << 3;
    v16bf afrag[4];
    #pragma unroll
    for (int kk = 0; kk < 4; ++kk) {
        const v8bf* p0 = (const v8bf*)&sH[arow][kk * 32 + koff];
        const v8bf* p1 = (const v8bf*)&sH[arow][kk * 32 + 16 + koff];
        afrag[kk] = __builtin_shufflevector(p0[0], p1[0],
            0, 1, 2, 3, 4, 5, 6, 7, 8, 9, 10, 11, 12, 13, 14, 15);
    }

    float m[8], den[8], num[8];
    #pragma unroll
    for (int r = 0; r < 8; ++r) { m[r] = -1e30f; den[r] = 0.f; num[r] = 0.f; }

    const int bn = lane & 15;
    const int bkoff = (lane >> 4) << 3;

    for (int sc = 0; sc < CHUNKK / SUBK; ++sc) {
        __syncthreads();
        const int k0 = kChunk0 + sc * SUBK;
        // stage keys sub-chunk: 64x128 fp32 -> bf16 in LDS (2048 float4)
        #pragma unroll
        for (int i = 0; i < 8; ++i) {
            int idx = t + i * 256;
            int r = idx >> 5, c = (idx & 31) << 2;
            float4 g = *(const float4*)&keys[(size_t)(k0 + r) * KD + c];
            uint2 pk;
            pk.x = bf16pk(g.x, g.y);
            pk.y = bf16pk(g.z, g.w);
            *(uint2*)&sK[r][c] = pk;
        }
        if (t < SUBK) sV[t] = values[k0 + t];
        __syncthreads();

        #pragma unroll
        for (int nt = 0; nt < 4; ++nt) {
            const int n = (nt << 4) + bn;
            v8f c = {0.f, 0.f, 0.f, 0.f, 0.f, 0.f, 0.f, 0.f};
            #pragma unroll
            for (int kk = 0; kk < 4; ++kk) {
                const v8bf* q0 = (const v8bf*)&sK[n][kk * 32 + bkoff];
                const v8bf* q1 = (const v8bf*)&sK[n][kk * 32 + 16 + bkoff];
                v16bf bfrag = __builtin_shufflevector(q0[0], q1[0],
                    0, 1, 2, 3, 4, 5, 6, 7, 8, 9, 10, 11, 12, 13, 14, 15);
                c = __builtin_amdgcn_wmma_f32_16x16x32_bf16(
                        false, afrag[kk], false, bfrag, (short)0, c, false, false);
            }
            const float v = sV[(nt << 4) + bn];
            #pragma unroll
            for (int r = 0; r < 8; ++r) {   // scores already in log2 domain
                float z  = c[r];
                float mn = fmaxf(m[r], z);
                float s  = EXP2F(m[r] - mn);
                float e  = EXP2F(z - mn);
                den[r] = fmaf(den[r], s, e);
                num[r] = fmaf(num[r], s, e * v);
                m[r] = mn;
            }
        }
    }

    // reduce (m,den,num) across the 16 lanes that share each row
    #pragma unroll
    for (int r = 0; r < 8; ++r) {
        float mm = m[r], dd = den[r], nn = num[r];
        #pragma unroll
        for (int mask = 1; mask <= 8; mask <<= 1) {
            float mo = __shfl_xor(mm, mask, 32);
            float dm = __shfl_xor(dd, mask, 32);
            float nm = __shfl_xor(nn, mask, 32);
            lse_merge(mm, dd, nn, mo, dm, nm);
        }
        if ((lane & 15) == 0) {
            int row = mBase + (wave << 4) + r + ((lane >> 4) << 3);
            float* p = partials + ((size_t)row * NCHUNK + blockIdx.x) * 3;
            p[0] = mm; p[1] = dd; p[2] = nn;
        }
    }
}

// ---------------------------------------------------------------------------
// K3: merge chunk partials per row -> qs = num/den
// ---------------------------------------------------------------------------
__global__ __launch_bounds__(32)
void k_reduce(const float* __restrict__ partials, float* __restrict__ out) {
    const int row = blockIdx.x, lane = threadIdx.x;
    float m = -1e30f, d = 0.f, n = 0.f;
    for (int c = lane; c < NCHUNK; c += 32) {
        const float* p = partials + ((size_t)row * NCHUNK + c) * 3;
        lse_merge(m, d, n, p[0], p[1], p[2]);
    }
    #pragma unroll
    for (int mask = 16; mask >= 1; mask >>= 1) {
        float mo = __shfl_xor(m, mask, 32);
        float dm = __shfl_xor(d, mask, 32);
        float nm = __shfl_xor(n, mask, 32);
        lse_merge(m, d, n, mo, dm, nm);
    }
    if (lane == 0) out[row] = n / d;
}

extern "C" void kernel_launch(void* const* d_in, const int* in_sizes, int n_in,
                              void* d_out, int out_size, void* d_ws, size_t ws_size,
                              hipStream_t stream) {
    (void)in_sizes; (void)n_in; (void)out_size; (void)ws_size;
    const float* obs      = (const float*)d_in[0];
    const float* action   = (const float*)d_in[1];
    const float* pred_W   = (const float*)d_in[2];
    const float* pred_b   = (const float*)d_in[3];
    const float* t1_W     = (const float*)d_in[4];
    const float* t1_b     = (const float*)d_in[5];
    const float* t2_W     = (const float*)d_in[6];
    const float* t2_b     = (const float*)d_in[7];
    const float* keys     = (const float*)d_in[8];
    const float* values   = (const float*)d_in[9];
    const float* log_temp = (const float*)d_in[10];
    float* out = (float*)d_out;

    unsigned short* hb = (unsigned short*)d_ws;                 // 1024*128*2 = 256 KB
    float* partials = (float*)((char*)d_ws + (size_t)BATCH * KD * 2); // 1024*128*3*4 = 1.5 MB

    k_trunk<<<BATCH, 128, 0, stream>>>(obs, action, pred_W, pred_b,
                                       t1_W, t1_b, t2_W, t2_b, log_temp, hb);
    k_scores<<<dim3(NCHUNK, BATCH / MTILE), 256, 0, stream>>>(keys, values, hb, partials);
    k_reduce<<<BATCH, 32, 0, stream>>>(partials, out);
}
